// RoPEAttention_90039694393600
// MI455X (gfx1250) — matrix operand order repaired and checked by reading, not compiled, not run on hardware
//
#include <hip/hip_runtime.h>

// ---------------- problem constants ----------------
#define BB   16
#define NN   577
#define CC   768
#define HH   12
#define DD   64
#define TC   2304          // 3*C
#define MM   (BB * NN)     // 9232 rows of the two big GEMMs
#define NKPAD 640          // padded key dim for V^T (>= 10*64, 16B-aligned rows)

typedef __attribute__((ext_vector_type(16))) __bf16 v16bf;
typedef __attribute__((ext_vector_type(8)))  float  v8f;

union Frag16 { v16bf v; uint4 q[2]; };

__device__ __forceinline__ v8f v8zero() {
  v8f z = {0.f, 0.f, 0.f, 0.f, 0.f, 0.f, 0.f, 0.f};
  return z;
}

// A-matrix fragment (16x32 bf16). Lane l holds row (l&15); K chunks at
// (l>>4)*8 .. +7 and +16 .. +23  -> two contiguous 16B loads.
__device__ __forceinline__ v16bf wload_a(const __bf16* base, int ld,
                                         int row0, int k0, int lane, int rowmax) {
  int mr = lane & 15, hf = lane >> 4;
  int row = row0 + mr; row = row < rowmax ? row : rowmax;
  const __bf16* p = base + (size_t)row * ld + k0 + hf * 8;
  Frag16 f;
  f.q[0] = *(const uint4*)(p);
  f.q[1] = *(const uint4*)(p + 16);
  return f.v;
}

// B-matrix fragment (32x16 bf16) read from a K-contiguous (N,K) layout.
// Lane l holds column (l&15); K chunks at (l>>4)*8 .. +7 and +16 .. +23 -> two 16B loads.
__device__ __forceinline__ v16bf wload_bt(const __bf16* base, int ld,
                                          int n0, int k0, int lane, int rowmax) {
  int col = lane & 15, kb = (lane >> 4) * 8;
  int row = n0 + col; row = row < rowmax ? row : rowmax;
  const __bf16* p = base + (size_t)row * ld + k0 + kb;
  Frag16 f;
  f.q[0] = *(const uint4*)(p);
  f.q[1] = *(const uint4*)(p + 16);
  return f.v;
}

__device__ __forceinline__ v8f wmma_bf16(v16bf a, v16bf b, v8f c) {
  return __builtin_amdgcn_wmma_f32_16x16x32_bf16(false, a, false, b,
                                                 (short)0, c, false, false);
}

// ---------------- f32 -> bf16 convert (x) ----------------
__global__ __launch_bounds__(256) void cvt_bf16_kernel(const float* __restrict__ src,
                                                       __bf16* __restrict__ dst, int n4) {
  int i = blockIdx.x * blockDim.x + threadIdx.x;
  if (i < n4) {
    float4 f = ((const float4*)src)[i];
    union { __bf16 h[4]; uint2 u; } o;
    o.h[0] = (__bf16)f.x; o.h[1] = (__bf16)f.y;
    o.h[2] = (__bf16)f.z; o.h[3] = (__bf16)f.w;
    ((uint2*)dst)[i] = o.u;
  }
}

// ---------------- weight transpose + convert: (K,N) f32 -> (N,K) bf16 ----------------
__global__ __launch_bounds__(256) void prep_w_kernel(const float* __restrict__ w,
                                                     __bf16* __restrict__ wt,
                                                     int K, int Nn) {
  int tid = blockIdx.x * blockDim.x + threadIdx.x;
  int total = K * Nn;
  if (tid < total) {
    int k = tid % K;          // k fastest -> coalesced bf16 writes
    int n = tid / K;
    wt[(size_t)n * K + k] = (__bf16)w[(size_t)k * Nn + n];
  }
}

// ---------------- generic GEMM: C(M,Nn) = A(M,K)bf16 @ BT(Nn,K)bf16 + bias ----------------
// mode 0: QKV epilogue -> scatter to qh/kh (B,H,N,D) and vT (B,H,D,NKPAD), bf16
// mode 1: plain f32 store to outF (row stride Nn)
// WG = 128 threads = 4 waves in a 2x2 grid; WG tile 128x64; wave tile 64x32.
__global__ __launch_bounds__(128) void gemm_bf16_kernel(
    const __bf16* __restrict__ A, const __bf16* __restrict__ BT,
    const float* __restrict__ bias, int M, int Nn, int K, int mode,
    float* __restrict__ outF, __bf16* __restrict__ qh,
    __bf16* __restrict__ kh, __bf16* __restrict__ vT) {
  int wid  = threadIdx.x >> 5;
  int lane = threadIdx.x & 31;
  int m0 = blockIdx.x * 128 + (wid & 1) * 64;
  int n0 = blockIdx.y * 64  + (wid >> 1) * 32;

  v8f acc[4][2];
#pragma unroll
  for (int mi = 0; mi < 4; ++mi)
#pragma unroll
    for (int ni = 0; ni < 2; ++ni) acc[mi][ni] = v8zero();

  for (int k0 = 0; k0 < K; k0 += 32) {
    v16bf a[4], b[2];
#pragma unroll
    for (int mi = 0; mi < 4; ++mi)
      a[mi] = wload_a(A, K, m0 + mi * 16, k0, lane, M - 1);
#pragma unroll
    for (int ni = 0; ni < 2; ++ni)
      b[ni] = wload_bt(BT, K, n0 + ni * 16, k0, lane, Nn - 1);
    if (k0 + 32 < K) {  // software prefetch of next K-slab (global_prefetch_b8)
      __builtin_prefetch(A + (size_t)(m0 + (lane & 15)) * K + k0 + 32, 0, 1);
      __builtin_prefetch(BT + (size_t)(n0 + (lane & 15)) * K + k0 + 32, 0, 1);
    }
#pragma unroll
    for (int mi = 0; mi < 4; ++mi)
#pragma unroll
      for (int ni = 0; ni < 2; ++ni)
        acc[mi][ni] = wmma_bf16(a[mi], b[ni], acc[mi][ni]);
  }

  // epilogue: C layout -> lane holds col = lane&15, rows r + 8*(lane>>4)
  int col_lo = lane & 15, hf = lane >> 4;
#pragma unroll
  for (int ni = 0; ni < 2; ++ni) {
    int col = n0 + ni * 16 + col_lo;
    float bv = bias[col];
#pragma unroll
    for (int mi = 0; mi < 4; ++mi) {
#pragma unroll
      for (int r = 0; r < 8; ++r) {
        int row = m0 + mi * 16 + r + 8 * hf;
        if (row < M) {
          float val = acc[mi][ni][r] + bv;
          if (mode == 1) {
            outF[(size_t)row * Nn + col] = val;
          } else {
            int t = col / 768, rem = col - t * 768;
            int h = rem >> 6, d = rem & 63;
            int bI = row / NN, n = row - bI * NN;
            __bf16 vb = (__bf16)val;
            if (t == 0)      qh[(((size_t)bI * HH + h) * NN + n) * DD + d] = vb;
            else if (t == 1) kh[(((size_t)bI * HH + h) * NN + n) * DD + d] = vb;
            else             vT[(((size_t)bI * HH + h) * DD + d) * NKPAD + n] = vb;
          }
        }
      }
    }
  }
}

// ---------------- RoPE (in place on q,k bf16) + fold 1/sqrt(D) into q ----------------
__global__ __launch_bounds__(256) void rope_kernel(__bf16* __restrict__ qh,
                                                   __bf16* __restrict__ kh,
                                                   const float* __restrict__ cosp,
                                                   const float* __restrict__ sinp) {
  int t = blockIdx.x * blockDim.x + threadIdx.x;
  int total = BB * HH * NN * (DD / 2);
  if (t >= total) return;
  int j = t & 31;
  int rest = t >> 5;
  int n = rest % NN;
  int bh = rest / NN;
  size_t base = ((size_t)bh * NN + n) * DD + 2 * j;
  const float scale = 0.125f;  // 64^-0.5
  float q0 = (float)qh[base], q1 = (float)qh[base + 1];
  if (n == 0) {
    qh[base] = (__bf16)(q0 * scale);
    qh[base + 1] = (__bf16)(q1 * scale);
    return;  // CLS token: no rotation, k unchanged
  }
  float k0 = (float)kh[base], k1 = (float)kh[base + 1];
  float c = cosp[(size_t)(n - 1) * 32 + j];
  float s = sinp[(size_t)(n - 1) * 32 + j];
  qh[base]     = (__bf16)((q0 * c - q1 * s) * scale);
  qh[base + 1] = (__bf16)((q0 * s + q1 * c) * scale);
  kh[base]     = (__bf16)(k0 * c - k1 * s);
  kh[base + 1] = (__bf16)(k0 * s + k1 * c);
}

// ---------------- flash attention: per-(b,h), 64-query tiles, 4 waves x 16 rows ----------------
__global__ __launch_bounds__(128) void attn_kernel(const __bf16* __restrict__ qh,
                                                   const __bf16* __restrict__ kh,
                                                   const __bf16* __restrict__ vT,
                                                   __bf16* __restrict__ ao) {
  __shared__ __align__(16) __bf16 Plds[4][16][72];  // per-wave P staging, padded rows

  int bh = blockIdx.x;                 // 0..191
  int bI = bh / HH, h = bh % HH;
  int wid  = threadIdx.x >> 5;
  int lane = threadIdx.x & 31;
  int mr = lane & 15, hf = lane >> 4, col = lane & 15;

  const __bf16* qb = qh + (size_t)bh * NN * DD;
  const __bf16* kb = kh + (size_t)bh * NN * DD;
  const __bf16* vb = vT + (size_t)bh * DD * NKPAD;

  int q0 = blockIdx.y * 64 + wid * 16;

  v16bf qa[2];
#pragma unroll
  for (int kk = 0; kk < 2; ++kk)
    qa[kk] = wload_a(qb, DD, q0, kk * 32, lane, NN - 1);

  v8f o[4];
#pragma unroll
  for (int jd = 0; jd < 4; ++jd) o[jd] = v8zero();
  float mrow[8], lrow[8];
#pragma unroll
  for (int r = 0; r < 8; ++r) { mrow[r] = -1e30f; lrow[r] = 0.f; }

  for (int kt = 0; kt < 10; ++kt) {
    int kbase = kt * 64;

    // S = q @ k^T  (16 x 64 per wave)
    v8f s[4];
#pragma unroll
    for (int j = 0; j < 4; ++j) {
      s[j] = v8zero();
#pragma unroll
      for (int kk = 0; kk < 2; ++kk) {
        v16bf kf = wload_bt(kb, DD, kbase + j * 16, kk * 32, lane, NN - 1);
        s[j] = wmma_bf16(qa[kk], kf, s[j]);
      }
    }

    // mask invalid keys (column index is uniform over r for a lane)
#pragma unroll
    for (int j = 0; j < 4; ++j) {
      int key = kbase + j * 16 + col;
      if (key >= NN) {
#pragma unroll
        for (int r = 0; r < 8; ++r) s[j][r] = -1e30f;
      }
    }

    // online softmax per output row (row lives in one 16-lane half at vgpr r)
    float alpha[8];
#pragma unroll
    for (int r = 0; r < 8; ++r) {
      float mx = fmaxf(fmaxf(s[0][r], s[1][r]), fmaxf(s[2][r], s[3][r]));
#pragma unroll
      for (int off = 1; off < 16; off <<= 1)
        mx = fmaxf(mx, __shfl_xor(mx, off, 32));
      float mnew = fmaxf(mrow[r], mx);
      alpha[r] = __expf(mrow[r] - mnew);
      mrow[r] = mnew;
      float rs = 0.f;
#pragma unroll
      for (int j = 0; j < 4; ++j) {
        float pv = __expf(s[j][r] - mnew);
        s[j][r] = pv;
        rs += pv;
      }
#pragma unroll
      for (int off = 1; off < 16; off <<= 1)
        rs += __shfl_xor(rs, off, 32);
      lrow[r] = lrow[r] * alpha[r] + rs;
    }

    // rescale running O
#pragma unroll
    for (int jd = 0; jd < 4; ++jd)
#pragma unroll
      for (int r = 0; r < 8; ++r) o[jd][r] *= alpha[r];

    // P: C-frag layout -> A-frag layout via LDS (bf16)
    __syncthreads();
#pragma unroll
    for (int j = 0; j < 4; ++j)
#pragma unroll
      for (int r = 0; r < 8; ++r)
        Plds[wid][r + 8 * hf][j * 16 + col] = (__bf16)s[j][r];
    __syncthreads();

    v16bf pa[2];
#pragma unroll
    for (int kk = 0; kk < 2; ++kk) {
      const __bf16* p = &Plds[wid][mr][kk * 32 + hf * 8];
      Frag16 f;
      f.q[0] = *(const uint4*)(p);
      f.q[1] = *(const uint4*)(p + 16);
      pa[kk] = f.v;
    }

    // O += P @ V  (V^T rows are d, K-contiguous in key)
#pragma unroll
    for (int jd = 0; jd < 4; ++jd)
#pragma unroll
      for (int kk = 0; kk < 2; ++kk) {
        v16bf vf = wload_bt(vb, NKPAD, jd * 16, kbase + kk * 32, lane, DD - 1);
        o[jd] = wmma_bf16(pa[kk], vf, o[jd]);
      }
  }

  // normalize + store to (B, N, C) bf16 for the projection GEMM
#pragma unroll
  for (int r = 0; r < 8; ++r) {
    int n = q0 + r + 8 * hf;
    if (n < NN) {
      float inv = 1.0f / lrow[r];
#pragma unroll
      for (int jd = 0; jd < 4; ++jd)
        ao[((size_t)bI * NN + n) * CC + h * DD + jd * 16 + col] =
            (__bf16)(o[jd][r] * inv);
    }
  }
}

// ---------------- host launch ----------------
extern "C" void kernel_launch(void* const* d_in, const int* in_sizes, int n_in,
                              void* d_out, int out_size, void* d_ws, size_t ws_size,
                              hipStream_t stream) {
  const float* x      = (const float*)d_in[0];
  const float* cosp   = (const float*)d_in[1];
  const float* sinp   = (const float*)d_in[2];
  const float* w_qkv  = (const float*)d_in[3];
  const float* b_qkv  = (const float*)d_in[4];
  const float* w_proj = (const float*)d_in[5];
  const float* b_proj = (const float*)d_in[6];
  float* out = (float*)d_out;

  char* ws = (char*)d_ws;
  size_t off = 0;
  auto carve = [&](size_t bytes) -> char* {
    char* p = ws + off;
    off = (off + bytes + 255) & ~(size_t)255;
    return p;
  };
  __bf16* xb     = (__bf16*)carve((size_t)MM * CC * 2);        // x in bf16
  __bf16* wqkvT  = (__bf16*)carve((size_t)TC * CC * 2);        // (3C, C) bf16
  __bf16* wprojT = (__bf16*)carve((size_t)CC * CC * 2);        // (C, C) bf16
  __bf16* qh     = (__bf16*)carve((size_t)BB * HH * NN * DD * 2);
  __bf16* kh     = (__bf16*)carve((size_t)BB * HH * NN * DD * 2);
  __bf16* vT     = (__bf16*)carve((size_t)BB * HH * DD * NKPAD * 2);
  __bf16* ao     = (__bf16*)carve((size_t)MM * CC * 2);        // attn out (B,N,C) bf16
  (void)ws_size; (void)in_sizes; (void)n_in; (void)out_size;

  // 1) convert x to bf16
  {
    int n4 = (MM * CC) / 4;
    cvt_bf16_kernel<<<(n4 + 255) / 256, 256, 0, stream>>>(x, xb, n4);
  }
  // 2) transpose+convert weights to (N,K) bf16
  prep_w_kernel<<<(CC * TC + 255) / 256, 256, 0, stream>>>(w_qkv, wqkvT, CC, TC);
  prep_w_kernel<<<(CC * CC + 255) / 256, 256, 0, stream>>>(w_proj, wprojT, CC, CC);
  // 3) QKV GEMM with head-scatter epilogue
  {
    dim3 grid((MM + 127) / 128, TC / 64);
    gemm_bf16_kernel<<<grid, 128, 0, stream>>>(xb, wqkvT, b_qkv, MM, TC, CC, 0,
                                               nullptr, qh, kh, vT);
  }
  // 4) RoPE (in place, scale folded into q)
  {
    int total = BB * HH * NN * (DD / 2);
    rope_kernel<<<(total + 255) / 256, 256, 0, stream>>>(qh, kh, cosp, sinp);
  }
  // 5) flash attention
  {
    dim3 grid(BB * HH, (NN + 63) / 64);
    attn_kernel<<<grid, 128, 0, stream>>>(qh, kh, vT, ao);
  }
  // 6) output projection
  {
    dim3 grid((MM + 127) / 128, CC / 64);
    gemm_bf16_kernel<<<grid, 128, 0, stream>>>(ao, wprojT, b_proj, MM, CC, CC, 1,
                                               out, nullptr, nullptr, nullptr);
  }
}
